// ProductKeyMemory_83004537962930
// MI455X (gfx1250) — compile-verified, matrix operand order checked by reading, not run on hardware
//
#include <hip/hip_runtime.h>

// ---------------------------------------------------------------------------
// Problem constants (must match reference)
// ---------------------------------------------------------------------------
#define B_SZ      4096
#define DIM       2048
#define QDIM      512
#define KP        256
#define NUM_KEYS  1024
#define K_COARSE  4
#define NUM_CAND  64
#define CAP       8
#define INV_TEMP  (1.0f / 0.07f)

typedef __bf16 v16bf __attribute__((ext_vector_type(16)));
typedef float  v8f   __attribute__((ext_vector_type(8)));

static __device__ __forceinline__ __bf16 f32_to_bf16(float f) {
  unsigned u = __builtin_bit_cast(unsigned, f);
  unsigned r = u + 0x7FFFu + ((u >> 16) & 1u);   // round-to-nearest-even
  unsigned short h = (unsigned short)(r >> 16);
  return __builtin_bit_cast(__bf16, h);
}

// CDNA5 async global->LDS copy (per-lane 16B), tracked by ASYNCcnt.
static __device__ __forceinline__ void async_copy_b128(unsigned lds_off, const void* gaddr) {
  asm volatile("global_load_async_to_lds_b128 %0, %1, off"
               :: "v"(lds_off), "v"((unsigned long long)(size_t)gaddr)
               : "memory");
}
static __device__ __forceinline__ void wait_async0() {
  asm volatile("s_wait_asynccnt 0x0" ::: "memory");
}

// ---------------------------------------------------------------------------
// f32 -> bf16 conversion pre-pass (grid-stride)
// ---------------------------------------------------------------------------
__global__ __launch_bounds__(256) void cvt_bf16_kernel(
    const float* __restrict__ src, __bf16* __restrict__ dst, int n) {
  int i = blockIdx.x * blockDim.x + threadIdx.x;
  int stride = gridDim.x * blockDim.x;
  for (; i < n; i += stride) dst[i] = f32_to_bf16(src[i]);
}

// ---------------------------------------------------------------------------
// q = embedding @ Wq^T   (4096x2048 @ 2048x512) -> qraw f32 [4096][512]
// One block = 16 rows x 512 cols; 8 waves, each wave 16x64 (4 WMMA n-tiles).
// A tile is double-buffered in LDS: next tile's f32 loads issue before the
// WMMA burst, cvt+ds_store land after it; one barrier per K step.
// ---------------------------------------------------------------------------
__global__ __launch_bounds__(256) void qproj_kernel(
    const float* __restrict__ emb, const __bf16* __restrict__ wqb,
    float* __restrict__ qraw) {
  __shared__ __align__(32) __bf16 As[2][16 * 32];

  const int tid   = threadIdx.x;
  const int lane  = tid & 31;
  const int wave  = tid >> 5;            // 0..7
  const int m0    = blockIdx.x * 16;
  const int n0    = wave * 64;
  const int arow  = lane & 15;           // A row / B column selector
  const int khalf = (lane >> 4) * 16;    // K half selector

  // staging slot for this thread: 2 elems of the 16x32 tile
  const int e = tid * 2;
  const int r = e >> 5;
  const int c = e & 31;
  const float* erow = emb + (m0 + r) * DIM + c;

  v8f acc[4] = {v8f{0}, v8f{0}, v8f{0}, v8f{0}};

  // prologue: stage K tile 0 into buffer 0
  As[0][r * 32 + c]     = f32_to_bf16(erow[0]);
  As[0][r * 32 + c + 1] = f32_to_bf16(erow[1]);
  __syncthreads();

  int p = 0;
  for (int k0 = 0; k0 < DIM; k0 += 32) {
    const bool has_next = (k0 + 32) < DIM;
    float f0 = 0.0f, f1 = 0.0f;
    if (has_next) {                      // issue next-tile loads early
      const float* ep = erow + k0 + 32;
      __builtin_prefetch(ep + 32, 0, 1); // tile after that -> global_prefetch_b8
      f0 = ep[0];
      f1 = ep[1];
    }

    // B fragments + WMMA burst on current buffer
    v16bf bf[4];
#pragma unroll
    for (int t = 0; t < 4; ++t)
      bf[t] = *(const v16bf*)(wqb + (n0 + t * 16 + arow) * DIM + k0 + khalf);
    v16bf a = *(const v16bf*)(&As[p][arow * 32 + khalf]);
#pragma unroll
    for (int t = 0; t < 4; ++t)
      acc[t] = __builtin_amdgcn_wmma_f32_16x16x32_bf16(
          false, a, false, bf[t], (short)0, acc[t], false, false);

    if (has_next) {                      // convert + stage into other buffer
      As[p ^ 1][r * 32 + c]     = f32_to_bf16(f0);
      As[p ^ 1][r * 32 + c + 1] = f32_to_bf16(f1);
    }
    __syncthreads();
    p ^= 1;
  }

  const int orow0 = m0 + ((lane >> 4) ? 8 : 0);
#pragma unroll
  for (int t = 0; t < 4; ++t) {
    const int col = n0 + t * 16 + arow;
#pragma unroll
    for (int i = 0; i < 8; ++i)
      qraw[(orow0 + i) * QDIM + col] = acc[t][i];
  }
}

// ---------------------------------------------------------------------------
// kraw[which] = keys[which] @ W[which]^T   (1024x256 @ 256x256) -> f32
// blockIdx.y: 0 = row path, 1 = col path.  4 waves x 64 cols = 256 cols.
// A tile (16x256 bf16, contiguous 8KB) staged via async global->LDS copy.
// K loop fully unrolled (8 steps x 4 WMMA) for load/WMMA pipelining.
// ---------------------------------------------------------------------------
__global__ __launch_bounds__(128) void keys_gemm_kernel(
    const __bf16* __restrict__ keysb,   // [2][1024][256]
    const __bf16* __restrict__ wrcb,    // [2][256][256]
    float* __restrict__ kraw) {         // [2][1024][256]
  __shared__ __align__(32) __bf16 Atile[16 * KP];   // 8 KB
  const int which = blockIdx.y;
  const __bf16* A  = keysb + which * (NUM_KEYS * KP);
  const __bf16* Bm = wrcb  + which * (KP * KP);
  float* out       = kraw  + which * (NUM_KEYS * KP);

  const int tid   = threadIdx.x;
  const int lane  = tid & 31;
  const int wave  = tid >> 5;           // 0..3
  const int m0    = blockIdx.x * 16;
  const int n0    = wave * 64;
  const int arow  = lane & 15;
  const int khalf = (lane >> 4) * 16;

  // async-stage contiguous 8KB A tile: 128 threads x 4 chunks x 16B
  {
    const unsigned lds_base = (unsigned)(size_t)(void*)Atile;
    const char* gbase = (const char*)(A + (size_t)m0 * KP);
#pragma unroll
    for (int rr = 0; rr < 4; ++rr) {
      const int byte_off = (tid + rr * 128) * 16;
      async_copy_b128(lds_base + byte_off, gbase + byte_off);
    }
  }
  wait_async0();
  __syncthreads();

  v8f acc[4] = {v8f{0}, v8f{0}, v8f{0}, v8f{0}};

#pragma unroll
  for (int k0 = 0; k0 < KP; k0 += 32) {
    v16bf bf[4];
#pragma unroll
    for (int t = 0; t < 4; ++t)
      bf[t] = *(const v16bf*)(Bm + (n0 + t * 16 + arow) * KP + k0 + khalf);
    v16bf a = *(const v16bf*)(&Atile[arow * KP + k0 + khalf]);
#pragma unroll
    for (int t = 0; t < 4; ++t)
      acc[t] = __builtin_amdgcn_wmma_f32_16x16x32_bf16(
          false, a, false, bf[t], (short)0, acc[t], false, false);
  }

  const int orow0 = m0 + ((lane >> 4) ? 8 : 0);
#pragma unroll
  for (int t = 0; t < 4; ++t) {
    const int col = n0 + t * 16 + arow;
#pragma unroll
    for (int i = 0; i < 8; ++i)
      out[(orow0 + i) * KP + col] = acc[t][i];
  }
}

// ---------------------------------------------------------------------------
// L2 normalize rows of length 256; write bf16 (and optional f32) outputs.
// One wave per row.
// ---------------------------------------------------------------------------
__global__ __launch_bounds__(256) void norm_kernel(
    const float* __restrict__ src, int src_stride,
    __bf16* __restrict__ dstb, float* __restrict__ dstf, int rows) {
  const int lane = threadIdx.x & 31;
  const int row  = blockIdx.x * (blockDim.x >> 5) + (threadIdx.x >> 5);
  if (row >= rows) return;

  const float* s = src + (size_t)row * src_stride;
  float vals[8];
  float ss = 0.0f;
#pragma unroll
  for (int i = 0; i < 8; ++i) {
    float f = s[lane + i * 32];
    vals[i] = f;
    ss += f * f;
  }
#pragma unroll
  for (int off = 16; off >= 1; off >>= 1) ss += __shfl_xor(ss, off, 32);
  const float nrm = sqrtf(ss);
  const float sc  = 1.0f / fmaxf(nrm, 1e-12f);
#pragma unroll
  for (int i = 0; i < 8; ++i) {
    float o = vals[i] * sc;
    dstb[(size_t)row * KP + lane + i * 32] = f32_to_bf16(o);
    if (dstf) dstf[(size_t)row * KP + lane + i * 32] = o;
  }
}

// ---------------------------------------------------------------------------
// scores[which] = qn[which] @ kn[which]^T / TEMP  (4096x256 @ 256x1024)
// One block = 16 rows x 1024 cols; 16 waves, each 16x64.
// A tile (16x256 bf16, contiguous 8KB) staged once via async global->LDS;
// K loop fully unrolled (8 steps x 4 WMMA) and barrier-free.
// ---------------------------------------------------------------------------
__global__ __launch_bounds__(512) void scores_kernel(
    const __bf16* __restrict__ qw,   // [2][4096][256]
    const __bf16* __restrict__ kw,   // [2][1024][256]
    float* __restrict__ scores) {    // [2][4096][1024]
  __shared__ __align__(32) __bf16 Qs[16 * KP];      // 8 KB
  const int which = blockIdx.y;
  const __bf16* Q = qw + (size_t)which * (B_SZ * KP);
  const __bf16* K = kw + (size_t)which * (NUM_KEYS * KP);
  float* out      = scores + (size_t)which * ((size_t)B_SZ * NUM_KEYS);

  const int tid   = threadIdx.x;
  const int lane  = tid & 31;
  const int wave  = tid >> 5;           // 0..15
  const int m0    = blockIdx.x * 16;
  const int n0    = wave * 64;
  const int arow  = lane & 15;
  const int khalf = (lane >> 4) * 16;

  // async-stage contiguous 8KB A tile: 512 threads x 16B
  {
    const unsigned lds_base = (unsigned)(size_t)(void*)Qs;
    const char* gbase = (const char*)(Q + (size_t)m0 * KP);
    async_copy_b128(lds_base + tid * 16, gbase + tid * 16);
  }
  wait_async0();
  __syncthreads();

  v8f acc[4] = {v8f{0}, v8f{0}, v8f{0}, v8f{0}};

#pragma unroll
  for (int k0 = 0; k0 < KP; k0 += 32) {
    v16bf bf[4];
#pragma unroll
    for (int t = 0; t < 4; ++t)
      bf[t] = *(const v16bf*)(K + (n0 + t * 16 + arow) * KP + k0 + khalf);
    v16bf a = *(const v16bf*)(&Qs[arow * KP + k0 + khalf]);
#pragma unroll
    for (int t = 0; t < 4; ++t)
      acc[t] = __builtin_amdgcn_wmma_f32_16x16x32_bf16(
          false, a, false, bf[t], (short)0, acc[t], false, false);
  }

  const int orow0 = m0 + ((lane >> 4) ? 8 : 0);
#pragma unroll
  for (int t = 0; t < 4; ++t) {
    const int col = n0 + t * 16 + arow;
#pragma unroll
    for (int i = 0; i < 8; ++i)
      out[(size_t)(orow0 + i) * NUM_KEYS + col] = acc[t][i] * INV_TEMP;
  }
}

// ---------------------------------------------------------------------------
// Per-row: top-4 of scores_1 and scores_2, grid cells, ragged gather with
// cap 8, order-preserving compaction, cyclic fill of 64 candidates.
// One wave per row; 8 waves per block.  Scan vectorized as float4.
// ---------------------------------------------------------------------------
__global__ __launch_bounds__(256) void topk_gather_kernel(
    const float* __restrict__ scores,   // [2][4096][1024]
    const int* __restrict__ offsets,
    const int* __restrict__ counts,
    const int* __restrict__ inv,
    int inv_len,
    int* __restrict__ cand) {
  __shared__ int bufs[8][NUM_CAND];
  const int lane = threadIdx.x & 31;
  const int wave = threadIdx.x >> 5;
  const int b    = blockIdx.x * 8 + wave;

  int tr[4], tc[4];
#pragma unroll
  for (int half = 0; half < 2; ++half) {
    const float* row =
        scores + (size_t)half * ((size_t)B_SZ * NUM_KEYS) + (size_t)b * NUM_KEYS;
    const float4* rowv = (const float4*)row;
    float v[4]  = {-3.4e38f, -3.4e38f, -3.4e38f, -3.4e38f};
    int   id[4] = {0, 0, 0, 0};
    for (int t = 0; t < NUM_KEYS / 128; ++t) {     // 8 x float4 per lane
      const int i4 = lane + 32 * t;
      const float4 s4 = rowv[i4];
      const float se[4] = {s4.x, s4.y, s4.z, s4.w};
#pragma unroll
      for (int e2 = 0; e2 < 4; ++e2) {
        const float s = se[e2];
        const int j = i4 * 4 + e2;
        if (s > v[3]) {
          v[3] = s; id[3] = j;
#pragma unroll
          for (int q = 3; q > 0; --q)
            if (v[q] > v[q - 1]) {
              float tv = v[q]; v[q] = v[q - 1]; v[q - 1] = tv;
              int   ti = id[q]; id[q] = id[q - 1]; id[q - 1] = ti;
            }
        }
      }
    }
    // merge sorted top-4 lists across all 32 lanes (xor butterfly)
#pragma unroll
    for (int off = 16; off >= 1; off >>= 1) {
      float ov[4]; int oid[4];
#pragma unroll
      for (int q = 0; q < 4; ++q) {
        ov[q]  = __shfl_xor(v[q],  off, 32);
        oid[q] = __shfl_xor(id[q], off, 32);
      }
      float mv[4]; int mid[4];
      int i = 0, j2 = 0;
#pragma unroll
      for (int q = 0; q < 4; ++q) {
        bool self = (v[i] > ov[j2]) || (v[i] == ov[j2] && id[i] <= oid[j2]);
        mv[q]  = self ? v[i]  : ov[j2];
        mid[q] = self ? id[i] : oid[j2];
        i += self ? 1 : 0;
        j2 += self ? 0 : 1;
      }
#pragma unroll
      for (int q = 0; q < 4; ++q) { v[q] = mv[q]; id[q] = mid[q]; }
    }
    if (half == 0) {
#pragma unroll
      for (int q = 0; q < 4; ++q) tr[q] = id[q];
    } else {
#pragma unroll
      for (int q = 0; q < 4; ++q) tc[q] = id[q];
    }
  }

  // 16 cells handled by lanes 0..15
  int cnt = 0, off0 = 0;
  if (lane < 16) {
    int cell = tr[lane >> 2] * NUM_KEYS + tc[lane & 3];
    off0 = offsets[cell];
    int c = counts[cell];
    cnt = c < CAP ? c : CAP;
  }
  // inclusive prefix scan over lanes 0..15
  int inc = cnt;
#pragma unroll
  for (int d = 1; d < 16; d <<= 1) {
    int n = __shfl_up(inc, d, 32);
    if (lane >= d) inc += n;
  }
  const int total = __shfl(inc, 15, 32);
  const int base  = inc - cnt;
  if (lane < 16) {
    for (int t = 0; t < cnt; ++t) {
      int g = off0 + t;
      g = g < inv_len ? g : inv_len - 1;
      bufs[wave][base + t] = inv[g];
    }
  }
  __syncthreads();

  int* oc = cand + (size_t)b * NUM_CAND;
  for (int t = lane; t < NUM_CAND; t += 32) {
    int vout = 0;
    if (total > 0) vout = bufs[wave][t % total];
    oc[t] = vout;
  }
}

// ---------------------------------------------------------------------------
// Host launcher
// ---------------------------------------------------------------------------
extern "C" void kernel_launch(void* const* d_in, const int* in_sizes, int n_in,
                              void* d_out, int out_size, void* d_ws, size_t ws_size,
                              hipStream_t stream) {
  const float* emb  = (const float*)d_in[0];
  const float* Wq   = (const float*)d_in[1];
  const float* Wrow = (const float*)d_in[2];
  const float* Wcol = (const float*)d_in[3];
  const float* rkey = (const float*)d_in[4];
  const float* ckey = (const float*)d_in[5];
  const int*   coff = (const int*)d_in[6];
  const int*   ccnt = (const int*)d_in[7];
  const int*   inv  = (const int*)d_in[8];
  const int inv_len = in_sizes[8];

  // d_out regions: candidates i32 [B][64], scores [2][B][1024] f32, q1 [B][256] f32
  int*   cand   = (int*)d_out;
  float* scores = (float*)d_out + (size_t)B_SZ * NUM_CAND;
  float* q1out  = scores + 2ull * B_SZ * NUM_KEYS;

  // workspace carve-up
  char* w = (char*)d_ws;
  float* qraw = (float*)w;  w += sizeof(float) * (size_t)B_SZ * QDIM;        // 8 MB
  float* kraw = (float*)w;  w += sizeof(float) * 2ull * NUM_KEYS * KP;       // 2 MB
  __bf16* qw    = (__bf16*)w;  w += sizeof(__bf16) * 2ull * B_SZ * KP;       // 4 MB
  __bf16* kw    = (__bf16*)w;  w += sizeof(__bf16) * 2ull * NUM_KEYS * KP;   // 1 MB
  __bf16* wqb   = (__bf16*)w;  w += sizeof(__bf16) * (size_t)QDIM * DIM;     // 2 MB
  __bf16* keysb = (__bf16*)w;  w += sizeof(__bf16) * 2ull * NUM_KEYS * KP;   // 1 MB
  __bf16* wrcb  = (__bf16*)w;  w += sizeof(__bf16) * 2ull * KP * KP;         // 256 KB

  // 1. one-time bf16 conversions of GEMM operands
  cvt_bf16_kernel<<<2048, 256, 0, stream>>>(Wq, wqb, QDIM * DIM);
  cvt_bf16_kernel<<<1024, 256, 0, stream>>>(rkey, keysb, NUM_KEYS * KP);
  cvt_bf16_kernel<<<1024, 256, 0, stream>>>(ckey, keysb + NUM_KEYS * KP, NUM_KEYS * KP);
  cvt_bf16_kernel<<<256, 256, 0, stream>>>(Wrow, wrcb, KP * KP);
  cvt_bf16_kernel<<<256, 256, 0, stream>>>(Wcol, wrcb + KP * KP, KP * KP);

  // 2. WMMA GEMMs
  qproj_kernel<<<dim3(B_SZ / 16), 256, 0, stream>>>(emb, wqb, qraw);
  keys_gemm_kernel<<<dim3(NUM_KEYS / 16, 2), 128, 0, stream>>>(keysb, wrcb, kraw);

  // 3. L2 norms (q1 also emitted as f32 output)
  norm_kernel<<<B_SZ / 8, 256, 0, stream>>>(qraw, QDIM, qw, q1out, B_SZ);
  norm_kernel<<<B_SZ / 8, 256, 0, stream>>>(qraw + KP, QDIM, qw + (size_t)B_SZ * KP, nullptr, B_SZ);
  norm_kernel<<<NUM_KEYS / 8, 256, 0, stream>>>(kraw, KP, kw, nullptr, NUM_KEYS);
  norm_kernel<<<NUM_KEYS / 8, 256, 0, stream>>>(kraw + (size_t)NUM_KEYS * KP, KP,
                                                kw + (size_t)NUM_KEYS * KP, nullptr, NUM_KEYS);

  // 4. score GEMMs (both matrices via blockIdx.y)
  scores_kernel<<<dim3(B_SZ / 16, 2), 512, 0, stream>>>(qw, kw, scores);

  // 5. top-k + ragged gather + cyclic candidate fill
  topk_gather_kernel<<<B_SZ / 8, 256, 0, stream>>>(scores, coff, ccnt, inv, inv_len, cand);
}
